// Original_52819507806494
// MI455X (gfx1250) — matrix operand order repaired and checked by reading, not compiled, not run on hardware
//
#include <hip/hip_runtime.h>
#include <hip/hip_bf16.h>

#define NUM_USERS 40000
#define NUM_ITEMS 60000
#define N_NODES   100000
#define N_EDGES   1280000
#define EMB       64
#define NEG_SLOPE 0.2f

typedef __attribute__((ext_vector_type(16))) _Float16 v16h;
typedef __attribute__((ext_vector_type(8)))  float    v8f;

// Weight-fragment store: per layer 16 fragments ((m*2+kc)*4+n0), each
// 32 lanes x 16 halves (B-matrix 32x16 f16 layout). 8192 halves per layer.
#define FRAGS_PER_LAYER 16
#define FRAG_HALVES     (32 * 16)
#define LAYER_HALVES    (FRAGS_PER_LAYER * FRAG_HALVES)   // 8192
#define TOTAL_HALVES    (3 * LAYER_HALVES)                // 24576

// ---------------------------------------------------------------------------
// init: e = concat(user_emb, item_emb); acc = e
// ---------------------------------------------------------------------------
__global__ __launch_bounds__(256) void ngcf_init(
    const float* __restrict__ ue, const float* __restrict__ ie,
    float* __restrict__ e, float* __restrict__ acc)
{
  int idx = blockIdx.x * blockDim.x + threadIdx.x;
  if (idx >= N_NODES * EMB) return;
  float v = (idx < NUM_USERS * EMB) ? ue[idx] : ie[idx - NUM_USERS * EMB];
  e[idx]   = v;
  acc[idx] = v;
}

// ---------------------------------------------------------------------------
// zero x
// ---------------------------------------------------------------------------
__global__ __launch_bounds__(256) void ngcf_zero(float* __restrict__ x)
{
  int idx = blockIdx.x * blockDim.x + threadIdx.x;
  if (idx < N_NODES * EMB) x[idx] = 0.0f;
}

// ---------------------------------------------------------------------------
// Pre-pack W1/W2 (all 3 layers) into f16 B-matrix fragments in the exact
// per-lane VGPR layout: element j of lane l in fragment (m,kc,n0) is
// W[K=kc*32 + (l<16?0:8) + (j&7) + (j>>3)*16][col = n0*16 + (l&15)].
// Layer kernel then loads each fragment with one 32-byte vector deref.
// ---------------------------------------------------------------------------
__global__ __launch_bounds__(256) void ngcf_pack_w(
    const float* __restrict__ W1, const float* __restrict__ W2,
    _Float16* __restrict__ wf)
{
  int tid = blockIdx.x * blockDim.x + threadIdx.x;
  if (tid >= TOTAL_HALVES) return;
  int j     = tid & 15;
  int lane  = (tid >> 4) & 31;
  int g     = tid >> 9;        // global fragment id
  int n0    = g & 3;
  int kc    = (g >> 2) & 1;
  int m     = (g >> 3) & 1;
  int layer = g >> 4;
  int K   = kc * 32 + ((lane < 16) ? 0 : 8) + (j & 7) + ((j >> 3) << 4);
  int col = n0 * 16 + (lane & 15);
  const float* W = m ? W2 : W1;
  wf[tid] = (_Float16)W[layer * EMB * EMB + K * EMB + col];
}

// ---------------------------------------------------------------------------
// SpMM: one wave32 per edge; lane handles columns {2*lane, 2*lane+1}.
// 25.6 MB feature matrix is L2-resident on MI455X (192 MB L2); gathers and
// fp32 atomics resolve in L2.
// ---------------------------------------------------------------------------
__global__ __launch_bounds__(256) void ngcf_spmm(
    const int* __restrict__ erow, const int* __restrict__ ecol,
    const float* __restrict__ eval, const float* __restrict__ feats,
    float* __restrict__ x)
{
  int wave = (blockIdx.x * blockDim.x + threadIdx.x) >> 5;
  if (wave >= N_EDGES) return;
  int lane = threadIdx.x & 31;
  int r = erow[wave];
  int c = ecol[wave];
  float v = eval[wave];
  const float2 f = *(const float2*)(feats + c * EMB + lane * 2);
  float* dst = x + r * EMB + lane * 2;
  unsafeAtomicAdd(dst,     v * f.x);
  unsafeAtomicAdd(dst + 1, v * f.y);
}

// ---------------------------------------------------------------------------
// Fused layer: h = (e+x)@W1 + (x*e)@W2 + (b1+b2); leaky-relu; L2 row norm;
// e <- h_norm; acc += h_norm.
// One wave per 16-row block; 16x v_wmma_f32_16x16x32_f16, f32 accumulate.
// ---------------------------------------------------------------------------
__global__ __launch_bounds__(256) void ngcf_layer(
    const float* __restrict__ e_in, const float* __restrict__ x_in,
    const _Float16* __restrict__ wf,   // this layer's packed fragments
    const float* __restrict__ b1, const float* __restrict__ b2,
    float* __restrict__ e_out, float* __restrict__ acc)
{
  int wave = (blockIdx.x * blockDim.x + threadIdx.x) >> 5;
  if (wave >= N_NODES / 16) return;          // wave-uniform exit: EXEC all-1s
  int lane = threadIdx.x & 31;
  int hlf  = lane >> 4;
  int r0   = wave * 16;
  int M    = lane & 15;                      // A-matrix row for this lane
  int N0   = lane & 15;                      // B/C column within tile

  const float* erow = e_in + (r0 + M) * EMB;
  const float* xrow = x_in + (r0 + M) * EMB;

  // A fragments (16x32 f16 layout): per lane two contiguous runs of 8 K's.
  v16h aS[2], aP[2];
  #pragma unroll
  for (int kc = 0; kc < 2; ++kc) {
    int kb = kc * 32 + hlf * 8;
    float4 e0 = *(const float4*)(erow + kb);
    float4 e1 = *(const float4*)(erow + kb + 4);
    float4 e2 = *(const float4*)(erow + kb + 16);
    float4 e3 = *(const float4*)(erow + kb + 20);
    float4 x0 = *(const float4*)(xrow + kb);
    float4 x1 = *(const float4*)(xrow + kb + 4);
    float4 x2 = *(const float4*)(xrow + kb + 16);
    float4 x3 = *(const float4*)(xrow + kb + 20);
    float ev[16] = {e0.x,e0.y,e0.z,e0.w, e1.x,e1.y,e1.z,e1.w,
                    e2.x,e2.y,e2.z,e2.w, e3.x,e3.y,e3.z,e3.w};
    float xv[16] = {x0.x,x0.y,x0.z,x0.w, x1.x,x1.y,x1.z,x1.w,
                    x2.x,x2.y,x2.z,x2.w, x3.x,x3.y,x3.z,x3.w};
    #pragma unroll
    for (int i = 0; i < 16; ++i) {
      aS[kc][i] = (_Float16)(ev[i] + xv[i]);
      aP[kc][i] = (_Float16)(ev[i] * xv[i]);
    }
  }

  float cout[4][8];
  #pragma unroll
  for (int n0 = 0; n0 < 4; ++n0) {
    v8f c = {};
    #pragma unroll
    for (int kc = 0; kc < 2; ++kc) {
      // fragment id = (m*2 + kc)*4 + n0; 32B per lane, single vector deref
      v16h bw1 = *(const v16h*)(wf + ((0 * 2 + kc) * 4 + n0) * FRAG_HALVES + lane * 16);
      v16h bw2 = *(const v16h*)(wf + ((1 * 2 + kc) * 4 + n0) * FRAG_HALVES + lane * 16);
      c = __builtin_amdgcn_wmma_f32_16x16x32_f16(false, aS[kc], false, bw1,
                                                 (short)0, c, false, false);
      c = __builtin_amdgcn_wmma_f32_16x16x32_f16(false, aP[kc], false, bw2,
                                                 (short)0, c, false, false);
    }
    float bias = b1[n0 * 16 + N0] + b2[n0 * 16 + N0];
    #pragma unroll
    for (int v = 0; v < 8; ++v) {
      float h = c[v] + bias;
      cout[n0][v] = (h >= 0.0f) ? h : NEG_SLOPE * h;
    }
  }

  // Row L2 norms: C layout => VGPR v holds row M = v + 8*hlf across 16 lanes.
  #pragma unroll
  for (int v = 0; v < 8; ++v) {
    float s = 0.0f;
    #pragma unroll
    for (int n0 = 0; n0 < 4; ++n0) s += cout[n0][v] * cout[n0][v];
    s += __shfl_xor(s, 1);
    s += __shfl_xor(s, 2);
    s += __shfl_xor(s, 4);
    s += __shfl_xor(s, 8);                   // reduce within 16-lane half
    float inv = 1.0f / fmaxf(sqrtf(s), 1e-12f);
    int row = r0 + v + hlf * 8;
    #pragma unroll
    for (int n0 = 0; n0 < 4; ++n0) {
      float val = cout[n0][v] * inv;
      int idx = row * EMB + n0 * 16 + N0;
      e_out[idx] = val;
      acc[idx]  += val;
    }
  }
}

// ---------------------------------------------------------------------------
// final: out = acc / 4 (in place, acc aliases d_out)
// ---------------------------------------------------------------------------
__global__ __launch_bounds__(256) void ngcf_final(float* __restrict__ acc)
{
  int idx = blockIdx.x * blockDim.x + threadIdx.x;
  if (idx < N_NODES * EMB) acc[idx] *= 0.25f;
}

extern "C" void kernel_launch(void* const* d_in, const int* in_sizes, int n_in,
                              void* d_out, int out_size, void* d_ws, size_t ws_size,
                              hipStream_t stream)
{
  const int*   erow = (const int*)d_in[0];
  const int*   ecol = (const int*)d_in[1];
  const float* eval = (const float*)d_in[2];
  const float* ue   = (const float*)d_in[3];
  const float* ie   = (const float*)d_in[4];
  const float* W1   = (const float*)d_in[5];
  const float* b1   = (const float*)d_in[6];
  const float* W2   = (const float*)d_in[7];
  const float* b2   = (const float*)d_in[8];

  float*     acc = (float*)d_out;                         // accumulator in d_out
  float*     e   = (float*)d_ws;                          // current embedding
  float*     x   = e + (size_t)N_NODES * EMB;             // SpMM result
  _Float16*  wf  = (_Float16*)(x + (size_t)N_NODES * EMB);// packed W fragments

  const int elems      = N_NODES * EMB;
  const int elem_grid  = (elems + 255) / 256;
  const int spmm_grid  = (N_EDGES * 32 + 255) / 256;
  const int layer_grid = ((N_NODES / 16) + 7) / 8;        // 8 waves / block
  const int pack_grid  = (TOTAL_HALVES + 255) / 256;

  ngcf_init<<<elem_grid, 256, 0, stream>>>(ue, ie, e, acc);
  ngcf_pack_w<<<pack_grid, 256, 0, stream>>>(W1, W2, wf);

  for (int i = 0; i < 3; ++i) {
    ngcf_zero<<<elem_grid, 256, 0, stream>>>(x);
    ngcf_spmm<<<spmm_grid, 256, 0, stream>>>(erow, ecol, eval, e, x);
    ngcf_layer<<<layer_grid, 256, 0, stream>>>(
        e, x,
        wf + (size_t)i * LAYER_HALVES,
        b1 + i * EMB, b2 + i * EMB,
        e, acc);
  }

  ngcf_final<<<elem_grid, 256, 0, stream>>>(acc);
}
